// RankingModel_25237227831808
// MI455X (gfx1250) — matrix operand order repaired and checked by reading, not compiled, not run on hardware
//
#include <hip/hip_runtime.h>
#include <hip/hip_bf16.h>

typedef __attribute__((ext_vector_type(16))) _Float16 v16h;
typedef __attribute__((ext_vector_type(8)))  float    v8f;
typedef __attribute__((ext_vector_type(4)))  float    v4f;

#define Bn 8
#define Rn 8192
#define Kn 64
#define Hn 32
#define Fn 1024              // C*D
#define Mtot (Bn*Rn)         // 65536 rows
#define NEGV (-1e9f)

// ---------------------------------------------------------------------------
// Kernel 0: pack W1/W2 (f32, row-major) into f16 WMMA B-fragment lane order.
// w1f layout: [t(2)][it(32)][lane(32)][j(16)]  (32x16 fragment per (t,it))
// w2f layout: [t(4)][lane(32)][j(16)]
// B 32x16 lane map: K=(j>>3)*16 + (lane>>4)*8 + (j&7), n=lane&15
// ---------------------------------------------------------------------------
__global__ void pack_frags(const float* __restrict__ W1, const float* __restrict__ W2,
                           _Float16* __restrict__ w1f, _Float16* __restrict__ w2f) {
    int i = blockIdx.x * blockDim.x + threadIdx.x;
    const int N1 = 2 * 32 * 32 * 16;       // 32768
    if (i < N1) {
        int j    = i & 15;
        int lane = (i >> 4) & 31;
        int it   = (i >> 9) & 31;
        int t    = i >> 14;
        int n  = lane & 15;
        int hi = lane >> 4;
        int k  = it * 32 + ((j >> 3) << 4) + hi * 8 + (j & 7);    // B fragment K map for 32x16
        w1f[i] = (_Float16)W1[(size_t)k * Hn + t * 16 + n];
    }
    int i2 = i - N1;
    if (i2 >= 0 && i2 < 4 * 32 * 16) {
        int j    = i2 & 15;
        int lane = (i2 >> 4) & 31;
        int t    = i2 >> 9;
        int n  = lane & 15;
        int hi = lane >> 4;
        int k  = ((j >> 3) << 4) + hi * 8 + (j & 7);
        w2f[i2] = (_Float16)W2[(size_t)k * Kn + t * 16 + n];
    }
}

// ---------------------------------------------------------------------------
// Kernel 1: fused 2-layer MLP with f16 WMMA, f32 accumulate.
// One wave (32 lanes) owns a 16-row tile. 8 waves / block. HBM-bound:
// each table element is loaded exactly once, coalesced b128.
// ---------------------------------------------------------------------------
#define WAVES 8
__global__ __launch_bounds__(WAVES * 32)
void mlp_wmma(const float* __restrict__ table,
              const _Float16* __restrict__ w1f, const float* __restrict__ b1,
              const _Float16* __restrict__ w2f, const float* __restrict__ b2,
              float* __restrict__ logits) {
    __shared__ float hsm[WAVES][16][33];   // layer-1 output, per-wave transpose buffer

    const int lane = threadIdx.x & 31;
    const int wave = threadIdx.x >> 5;
    const int n  = lane & 15;
    const int hi = lane >> 4;
    const int row0 = (blockIdx.x * WAVES + wave) * 16;

    v8f c0 = {};   // layer-1 cols 0..15
    v8f c1 = {};   // layer-1 cols 16..31

    // ---- Layer 1: [16,1024] x [1024,32], K-step 32 ----
    const float* arow = table + (size_t)(row0 + n) * Fn + hi * 8;
    #pragma unroll 2
    for (int it = 0; it < 32; ++it) {
        const float* ap = arow + it * 32;
        v4f x0 = *(const v4f*)(ap);
        v4f x1 = *(const v4f*)(ap + 4);
        v4f x2 = *(const v4f*)(ap + 16);
        v4f x3 = *(const v4f*)(ap + 20);
        v16h a;
        #pragma unroll
        for (int q = 0; q < 4; ++q) {
            a[q]      = (_Float16)x0[q];
            a[4 + q]  = (_Float16)x1[q];
            a[8 + q]  = (_Float16)x2[q];
            a[12 + q] = (_Float16)x3[q];
        }
        v16h b0  = *(const v16h*)(w1f + ((size_t)(0 * 32 + it) * 32 + lane) * 16);
        v16h b1v = *(const v16h*)(w1f + ((size_t)(1 * 32 + it) * 32 + lane) * 16);
        c0 = __builtin_amdgcn_wmma_f32_16x16x32_f16(false, a, false, b0,  (short)0, c0, false, false);
        c1 = __builtin_amdgcn_wmma_f32_16x16x32_f16(false, a, false, b1v, (short)0, c1, false, false);
    }

    // bias + ReLU, dump to LDS (D layout: lane holds col n, VGPR v holds row hi*8+v)
    float bias0 = b1[n];
    float bias1 = b1[16 + n];
    #pragma unroll
    for (int v = 0; v < 8; ++v) {
        int m = hi * 8 + v;
        hsm[wave][m][n]      = fmaxf(c0[v] + bias0, 0.0f);
        hsm[wave][m][16 + n] = fmaxf(c1[v] + bias1, 0.0f);
    }
    __asm volatile("s_wait_dscnt 0" ::: "memory");

    // ---- Layer 2: [16,32] x [32,64], single K=32 WMMA per 16-col tile ----
    const float* hrow = &hsm[wave][n][0];   // A lane map: m = lane&15
    v16h a2;
    #pragma unroll
    for (int j = 0; j < 16; ++j) {
        int K = ((j >> 3) << 4) + hi * 8 + (j & 7);
        a2[j] = (_Float16)hrow[K];
    }

    #pragma unroll
    for (int t2 = 0; t2 < 4; ++t2) {
        v16h bw = *(const v16h*)(w2f + ((size_t)(t2 * 32) + lane) * 16);
        v8f d = {};
        d = __builtin_amdgcn_wmma_f32_16x16x32_f16(false, a2, false, bw, (short)0, d, false, false);
        float bias = b2[t2 * 16 + n];
        #pragma unroll
        for (int v = 0; v < 8; ++v) {
            int m = hi * 8 + v;
            logits[(size_t)(row0 + m) * Kn + t2 * 16 + n] = fmaxf(d[v] + bias, 0.0f);
        }
    }
}

// ---------------------------------------------------------------------------
// Kernel 2: sequential capacity-masked gumbel-argmax scan. One wave per batch,
// 2 experts per lane, one-row prefetch. Critical path minimized:
//   - fused value+index reduction gives max AND argmax in 5 shfl steps
//   - softmax evaluated only at the winner: z_am == mx so e_am = 1 exactly,
//     y_am = 1/sum (true divide, as jax's exp/sum); losers are exactly 0
//     in the reference ((0+y)-y == 0), so counts/outputs match bit-for-bit.
// ---------------------------------------------------------------------------
__global__ __launch_bounds__(32)
void scan_seq(const float* __restrict__ logits, const float* __restrict__ gumbel,
              const int* __restrict__ capacity, float* __restrict__ out) {
    const int b = blockIdx.x;
    const int lane = threadIdx.x;
    const float cap = (float)capacity[0];
    float cnt0 = 0.0f, cnt1 = 0.0f;
    const size_t base = (size_t)b * Rn * Kn;

    float l0 = logits[base + lane],      l1 = logits[base + 32 + lane];
    float g0 = gumbel[base + lane],      g1 = gumbel[base + 32 + lane];

    for (int r = 0; r < Rn; ++r) {
        size_t idx = base + (size_t)r * Kn;
        // prefetch next row while we crunch this one
        float nl0 = 0.f, nl1 = 0.f, ng0 = 0.f, ng1 = 0.f;
        if (r + 1 < Rn) {
            size_t nx = idx + Kn;
            nl0 = logits[nx + lane]; nl1 = logits[nx + 32 + lane];
            ng0 = gumbel[nx + lane]; ng1 = gumbel[nx + 32 + lane];
        }

        float z0 = l0 + (cnt0 < cap ? 0.0f : NEGV) + g0;
        float z1 = l1 + (cnt1 < cap ? 0.0f : NEGV) + g1;

        // fused max + argmax (lowest index on ties)
        float v = z0; int i = lane;
        if (z1 > v) { v = z1; i = lane + 32; }
        #pragma unroll
        for (int s = 16; s > 0; s >>= 1) {
            float ov = __shfl_xor(v, s, 32);
            int   oi = __shfl_xor(i, s, 32);
            if (ov > v || (ov == v && oi < i)) { v = ov; i = oi; }
        }
        // v == row max; softmax denominator
        float e0 = __expf(z0 - v), e1 = __expf(z1 - v);
        float sum = e0 + e1;
        #pragma unroll
        for (int s = 16; s > 0; s >>= 1) sum += __shfl_xor(sum, s, 32);

        float y_am = 1.0f / sum;               // exp(0)/sum at the winner
        float p_am = (1.0f + y_am) - y_am;     // straight-through, ref FP order

        float p0 = (i == lane)      ? p_am : 0.0f;
        float p1 = (i == lane + 32) ? p_am : 0.0f;
        cnt0 += p0; cnt1 += p1;
        out[idx + lane]      = p0;
        out[idx + 32 + lane] = p1;

        l0 = nl0; l1 = nl1; g0 = ng0; g1 = ng1;
    }
}

// ---------------------------------------------------------------------------
extern "C" void kernel_launch(void* const* d_in, const int* in_sizes, int n_in,
                              void* d_out, int out_size, void* d_ws, size_t ws_size,
                              hipStream_t stream) {
    const float* table = (const float*)d_in[0];
    const float* W1    = (const float*)d_in[1];
    const float* b1    = (const float*)d_in[2];
    const float* W2    = (const float*)d_in[3];
    const float* b2    = (const float*)d_in[4];
    const float* gum   = (const float*)d_in[5];
    const int*   cap   = (const int*)  d_in[6];
    float* out = (float*)d_out;

    // workspace: [logits 16MB f32][w1 frags 64KB f16][w2 frags 4KB f16]
    float*    logits = (float*)d_ws;
    _Float16* w1f    = (_Float16*)((char*)d_ws + (size_t)Mtot * Kn * sizeof(float));
    _Float16* w2f    = w1f + 2 * 32 * 32 * 16;

    const int packN = 2 * 32 * 32 * 16 + 4 * 32 * 16;
    pack_frags<<<(packN + 255) / 256, 256, 0, stream>>>(W1, W2, w1f, w2f);

    const int tiles  = Mtot / 16;                 // 4096
    const int blocks = tiles / WAVES;             // 512
    mlp_wmma<<<blocks, WAVES * 32, 0, stream>>>(table, w1f, b1, w2f, b2, logits);

    scan_seq<<<Bn, 32, 0, stream>>>(logits, gum, cap, out);
}
